// Node_GRU_22368189678091
// MI455X (gfx1250) — matrix operand order repaired and checked by reading, not compiled, not run on hardware
//
#include <hip/hip_runtime.h>
#include <cstdint>

// ---------------------------------------------------------------------------
// Types for CDNA5 WMMA (wave32, v_wmma_f32_16x16x32_bf16)
// ---------------------------------------------------------------------------
typedef __bf16 bf16_t;
typedef __attribute__((ext_vector_type(16))) __bf16 v16bf;
typedef __attribute__((ext_vector_type(8)))  __bf16 v8bf;
typedef __attribute__((ext_vector_type(8)))  float  v8f;

#define HID   384
#define NB    256
#define MAXA  128
#define MAXP  32
#define NHEAD 6
#define DK    64

__device__ __forceinline__ v8f wmma_bf16(v16bf a, v16bf b, v8f c) {
  return __builtin_amdgcn_wmma_f32_16x16x32_bf16(false, a, false, b, (short)0, c,
                                                 false, false);
}

// A-matrix fragment (16x32 bf16), row-major source, lane holds row lane%16.
// Layout: K = e + 16*hi (contiguous).
__device__ __forceinline__ v16bf loadA(const bf16_t* base, long long ld, int kb) {
  const int lane = threadIdx.x & 31;
  const int r    = lane & 15;
  const int hi   = lane >> 4;
  const bf16_t* p = base + (size_t)r * (size_t)ld + kb + hi * 16;
  v8bf lo  = *reinterpret_cast<const v8bf*>(p);
  v8bf hi8 = *reinterpret_cast<const v8bf*>(p + 8);
  v16bf f;
#pragma unroll
  for (int e = 0; e < 8; ++e) { f[e] = lo[e]; f[e + 8] = hi8[e]; }
  return f;
}

// B-matrix fragment (32x16 bf16) from a W^T view: W is [N,K] row-major,
// B[k][n] = W[n][k]. Layout: lane holds column n=lane%16, K = e + 16*hi.
// => a single contiguous 32-byte load per lane.
__device__ __forceinline__ v16bf loadB_w(const bf16_t* base, long long ldw, int kb) {
  const int lane = threadIdx.x & 31;
  const int c    = lane & 15;
  const int hi   = lane >> 4;
  const bf16_t* p = base + (size_t)c * (size_t)ldw + kb + hi * 16;
  return *reinterpret_cast<const v16bf*>(p);
}

// ---------------------------------------------------------------------------
// Small helper kernels
// ---------------------------------------------------------------------------
__global__ void offsets_kernel(const int* __restrict__ a_sizes,
                               const int* __restrict__ p_sizes,
                               int* __restrict__ a_off, int* __restrict__ p_off) {
  if (threadIdx.x == 0) {
    int s = 0;
    for (int i = 0; i < NB; ++i) { a_off[i] = s; s += a_sizes[i]; }
    a_off[NB] = s;
  } else if (threadIdx.x == 1) {
    int s = 0;
    for (int i = 0; i < NB; ++i) { p_off[i] = s; s += p_sizes[i]; }
    p_off[NB] = s;
  }
}

__global__ __launch_bounds__(256) void f32_to_bf16_kernel(
    const float* __restrict__ src, bf16_t* __restrict__ dst, int n) {
  int i = blockIdx.x * 256 + threadIdx.x;
  if (i < n) dst[i] = (bf16_t)src[i];
}

// Ragged -> zero-padded [NB, maxn, HID], converted to bf16.
__global__ __launch_bounds__(256) void pad_kernel(
    const float* __restrict__ feat, const int* __restrict__ off,
    const int* __restrict__ sizes, bf16_t* __restrict__ out, int maxn) {
  long long idx = (long long)blockIdx.x * 256 + threadIdx.x;
  int c = (int)(idx % HID);
  long long row = idx / HID;
  int i = (int)(row % maxn);
  int b = (int)(row / maxn);
  float v = (i < sizes[b]) ? feat[((size_t)(off[b] + i)) * HID + c] : 0.0f;
  out[idx] = (bf16_t)v;
}

// ---------------------------------------------------------------------------
// Generic WMMA GEMM: C[m,n] = sum_k A[m,k]*W[n,k] + bias[n]
// One wave computes a 16x64 tile (4 accumulators) so each A fragment feeds
// 4 WMMAs: load:WMMA ratio 5:4 instead of 2:1.
// ---------------------------------------------------------------------------
__global__ __launch_bounds__(128) void gemm_bias_kernel(
    const bf16_t* __restrict__ A, long long lda,
    const bf16_t* __restrict__ W, int K,
    const float* __restrict__ bias,
    int Ngroups, int total_tiles,
    bf16_t* __restrict__ out_b, float* __restrict__ out_f, int ldc) {
  const int wid = threadIdx.x >> 5;
  const int gt  = blockIdx.x * (blockDim.x >> 5) + wid;
  if (gt >= total_tiles) return;          // wave-uniform; EXEC stays all-ones
  const int mt = gt / Ngroups;
  const int ng = gt % Ngroups;            // group of 4 consecutive 16-col tiles
  const bf16_t* Abase = A + (size_t)mt * 16 * (size_t)lda;
  const bf16_t* Wbase = W + (size_t)ng * 64 * (size_t)K;
  v8f acc[4] = {{}, {}, {}, {}};
  for (int kb = 0; kb < K; kb += 32) {
    v16bf a = loadA(Abase, lda, kb);
#pragma unroll
    for (int j = 0; j < 4; ++j)
      acc[j] = wmma_bf16(a, loadB_w(Wbase + (size_t)j * 16 * K, K, kb), acc[j]);
  }
  const int lane = threadIdx.x & 31, cn = lane & 15, hi = lane >> 4;
#pragma unroll
  for (int j = 0; j < 4; ++j) {
    const int n = ng * 64 + j * 16 + cn;
    const float bv = bias[n];
#pragma unroll
    for (int r = 0; r < 8; ++r) {
      const size_t m = (size_t)mt * 16 + r + hi * 8;
      float v = acc[j][r] + bv;
      if (out_b) out_b[m * (size_t)ldc + n] = (bf16_t)v;
      else       out_f[m * (size_t)ldc + n] = v;
    }
  }
}

// Output projection + bias + residual(a_pad gathered from a_feat), bf16 out.
// Same 16x64-per-wave blocking (HID=384 -> 6 N-groups).
__global__ __launch_bounds__(128) void gemm_wo_residual_kernel(
    const bf16_t* __restrict__ O, const bf16_t* __restrict__ Wo,
    const float* __restrict__ bo, const float* __restrict__ a_feat,
    const int* __restrict__ a_off, const int* __restrict__ a_sizes,
    bf16_t* __restrict__ H) {
  const int wid = threadIdx.x >> 5;
  const int gt  = blockIdx.x * 4 + wid;
  const int mt = gt / 6;
  const int ng = gt % 6;
  const bf16_t* Abase = O  + (size_t)mt * 16 * HID;
  const bf16_t* Wbase = Wo + (size_t)ng * 64 * HID;
  v8f acc[4] = {{}, {}, {}, {}};
  for (int kb = 0; kb < HID; kb += 32) {
    v16bf a = loadA(Abase, HID, kb);
#pragma unroll
    for (int j = 0; j < 4; ++j)
      acc[j] = wmma_bf16(a, loadB_w(Wbase + (size_t)j * 16 * HID, HID, kb), acc[j]);
  }
  const int lane = threadIdx.x & 31, cn = lane & 15, hi = lane >> 4;
#pragma unroll
  for (int j = 0; j < 4; ++j) {
    const int n = ng * 64 + j * 16 + cn;
    const float bv = bo[n];
#pragma unroll
    for (int r = 0; r < 8; ++r) {
      const int m = mt * 16 + r + hi * 8;
      const int b = m >> 7, i = m & 127;
      float res = (i < a_sizes[b]) ? a_feat[((size_t)(a_off[b] + i)) * HID + n] : 0.0f;
      H[(size_t)m * HID + n] = (bf16_t)(acc[j][r] + bv + res);
    }
  }
}

// ---------------------------------------------------------------------------
// Fused masked cross-attention per (batch, head, 16-row tile).
// S = QK^T/8 (4 WMMAs), mask, softmax in-register, P@V (4 WMMAs).
// ---------------------------------------------------------------------------
__global__ __launch_bounds__(128) void attention_kernel(
    const bf16_t* __restrict__ Qb, const bf16_t* __restrict__ Kb,
    const bf16_t* __restrict__ Vb, const int* __restrict__ a_sizes,
    const int* __restrict__ p_sizes, bf16_t* __restrict__ Ob) {
  __shared__ alignas(32) bf16_t plds[4][16][32];
  const int wid  = threadIdx.x >> 5;
  const int lane = threadIdx.x & 31;
  const int gw = blockIdx.x * 4 + wid;
  const int mt = gw & 7;
  const int h  = (gw >> 3) % NHEAD;
  const int b  = gw / (8 * NHEAD);
  const int asz = a_sizes[b];
  const int psz = p_sizes[b];
  const bf16_t* Qp = Qb + ((size_t)b * MAXA + mt * 16) * HID + h * DK;
  const bf16_t* Kp = Kb + (size_t)b * MAXP * HID + h * DK;
  const bf16_t* Vp = Vb + (size_t)b * MAXP * HID + h * DK;

  v8f s0 = {}, s1 = {};
#pragma unroll
  for (int kb = 0; kb < DK; kb += 32) {
    v16bf a  = loadA(Qp, HID, kb);
    s0 = wmma_bf16(a, loadB_w(Kp,            HID, kb), s0);
    s1 = wmma_bf16(a, loadB_w(Kp + 16 * HID, HID, kb), s1);
  }

  const int cn = lane & 15;
  const int hi = lane >> 4;
#pragma unroll
  for (int r = 0; r < 8; ++r) {
    const int m = mt * 16 + r + hi * 8;
    const bool rowok = m < asz;
    float v0 = s0[r] * 0.125f;                 // 1/sqrt(dk=64)
    float v1 = s1[r] * 0.125f;
    if (!rowok || cn >= psz)        v0 = -1e9f;
    if (!rowok || (cn + 16) >= psz) v1 = -1e9f;
    float mx = fmaxf(v0, v1);
#pragma unroll
    for (int o = 8; o >= 1; o >>= 1) mx = fmaxf(mx, __shfl_xor(mx, o, 32));
    float p0 = __expf(v0 - mx), p1 = __expf(v1 - mx);
    float sm = p0 + p1;
#pragma unroll
    for (int o = 8; o >= 1; o >>= 1) sm += __shfl_xor(sm, o, 32);
    const float inv = 1.0f / sm;               // all-masked row -> 1/32 each
    const int row = r + hi * 8;
    plds[wid][row][cn]      = (bf16_t)(p0 * inv);
    plds[wid][row][cn + 16] = (bf16_t)(p1 * inv);
  }

  // P (16x32) reloaded in A-layout from LDS; V is [32,64] K-major: K=e+16*hi.
  v16bf pa = loadA(&plds[wid][0][0], 32, 0);
#pragma unroll
  for (int n0 = 0; n0 < DK; n0 += 16) {
    v16bf bv;
#pragma unroll
    for (int e = 0; e < 16; ++e)
      bv[e] = Vp[(size_t)(e + hi * 16) * HID + n0 + cn];
    v8f oacc = {};
    oacc = wmma_bf16(pa, bv, oacc);
#pragma unroll
    for (int r = 0; r < 8; ++r) {
      const int m = mt * 16 + r + hi * 8;
      Ob[((size_t)b * MAXA + m) * HID + h * DK + n0 + cn] = (bf16_t)oacc[r];
    }
  }
}

// ---------------------------------------------------------------------------
// h0 = max_t h[b,t,:], initialize GRU states + pooled accumulators.
// ---------------------------------------------------------------------------
__global__ __launch_bounds__(256) void h0_init_kernel(
    const bf16_t* __restrict__ H, float* __restrict__ sf, bf16_t* __restrict__ sfb,
    float* __restrict__ sb, bf16_t* __restrict__ sbb,
    float* __restrict__ accf, float* __restrict__ accb) {
  const int idx = blockIdx.x * 256 + threadIdx.x;   // NB*HID
  const int b = idx / HID, d = idx % HID;
  float m = -3.0e38f;
  for (int t = 0; t < MAXA; ++t)
    m = fmaxf(m, (float)H[((size_t)b * MAXA + t) * HID + d]);
  sf[idx] = m;  sb[idx] = m;
  sfb[idx] = (bf16_t)m;  sbb[idx] = (bf16_t)m;
  accf[idx] = 0.0f;  accb[idx] = 0.0f;
}

// GRU gate math + pooled accumulation (t < p_sizes[b]).
__global__ __launch_bounds__(256) void gru_gate_kernel(
    const float* __restrict__ gi, const float* __restrict__ gh,
    float* __restrict__ h, bf16_t* __restrict__ hb, float* __restrict__ acc,
    const int* __restrict__ p_sizes, int t) {
  const int idx = blockIdx.x * 256 + threadIdx.x;
  const int b = idx / HID, d = idx % HID;
  const float* gib = gi + (size_t)b * (3 * HID);
  const float* ghb = gh + (size_t)b * (3 * HID);
  const float r = 1.0f / (1.0f + __expf(-(gib[d] + ghb[d])));
  const float z = 1.0f / (1.0f + __expf(-(gib[HID + d] + ghb[HID + d])));
  const float n = tanhf(gib[2 * HID + d] + r * ghb[2 * HID + d]);
  const float hp = h[idx];
  const float hn = (1.0f - z) * n + z * hp;
  h[idx] = hn;
  hb[idx] = (bf16_t)hn;
  if (t < p_sizes[b]) acc[idx] += hn;
}

__global__ __launch_bounds__(256) void final_kernel(
    const float* __restrict__ accf, const float* __restrict__ accb,
    const int* __restrict__ p_sizes, float* __restrict__ out) {
  const int idx = blockIdx.x * 256 + threadIdx.x;
  const int b = idx / HID, d = idx % HID;
  const float inv = 1.0f / (float)p_sizes[b];
  out[(size_t)b * (2 * HID) + d]       = accf[idx] * inv;
  out[(size_t)b * (2 * HID) + HID + d] = accb[idx] * inv;
}

// ---------------------------------------------------------------------------
// Host orchestration
// ---------------------------------------------------------------------------
extern "C" void kernel_launch(void* const* d_in, const int* in_sizes, int n_in,
                              void* d_out, int out_size, void* d_ws, size_t ws_size,
                              hipStream_t stream) {
  (void)in_sizes; (void)n_in; (void)out_size; (void)ws_size;
  const float* a_feat = (const float*)d_in[0];
  const float* p_feat = (const float*)d_in[1];
  const float* Wq = (const float*)d_in[2];  const float* bq = (const float*)d_in[3];
  const float* Wk = (const float*)d_in[4];  const float* bk = (const float*)d_in[5];
  const float* Wv = (const float*)d_in[6];  const float* bv = (const float*)d_in[7];
  const float* Wo = (const float*)d_in[8];  const float* bo = (const float*)d_in[9];
  const float* Wih_f = (const float*)d_in[10]; const float* Whh_f = (const float*)d_in[11];
  const float* bih_f = (const float*)d_in[12]; const float* bhh_f = (const float*)d_in[13];
  const float* Wih_b = (const float*)d_in[14]; const float* Whh_b = (const float*)d_in[15];
  const float* bih_b = (const float*)d_in[16]; const float* bhh_b = (const float*)d_in[17];
  const int* a_sizes = (const int*)d_in[18];
  const int* p_sizes = (const int*)d_in[19];
  float* out = (float*)d_out;

  char* ws = (char*)d_ws;
  size_t off = 0;
  auto alloc = [&](size_t bytes) -> void* {
    off = (off + 255) & ~(size_t)255;
    void* p = ws + off;
    off += bytes;
    return p;
  };

  const size_t NA = (size_t)NB * MAXA * HID;   // 12,582,912
  const size_t NP = (size_t)NB * MAXP * HID;   //  3,145,728
  int*    a_off  = (int*)alloc((NB + 1) * sizeof(int));
  int*    p_off  = (int*)alloc((NB + 1) * sizeof(int));
  bf16_t* a_pad  = (bf16_t*)alloc(NA * 2);
  bf16_t* p_pad  = (bf16_t*)alloc(NP * 2);
  bf16_t* Qb     = (bf16_t*)alloc(NA * 2);
  bf16_t* Kbuf   = (bf16_t*)alloc(NP * 2);
  bf16_t* Vbuf   = (bf16_t*)alloc(NP * 2);
  bf16_t* Obuf   = (bf16_t*)alloc(NA * 2);
  bf16_t* Hbuf   = (bf16_t*)alloc(NA * 2);
  bf16_t* Wqb = (bf16_t*)alloc((size_t)HID * HID * 2);
  bf16_t* Wkb = (bf16_t*)alloc((size_t)HID * HID * 2);
  bf16_t* Wvb = (bf16_t*)alloc((size_t)HID * HID * 2);
  bf16_t* Wob = (bf16_t*)alloc((size_t)HID * HID * 2);
  bf16_t* Wihf = (bf16_t*)alloc((size_t)3 * HID * HID * 2);
  bf16_t* Whhf = (bf16_t*)alloc((size_t)3 * HID * HID * 2);
  bf16_t* Wihb = (bf16_t*)alloc((size_t)3 * HID * HID * 2);
  bf16_t* Whhb = (bf16_t*)alloc((size_t)3 * HID * HID * 2);
  float*  sf   = (float*)alloc((size_t)NB * HID * 4);
  float*  sb   = (float*)alloc((size_t)NB * HID * 4);
  bf16_t* sfb  = (bf16_t*)alloc((size_t)NB * HID * 2);
  bf16_t* sbb  = (bf16_t*)alloc((size_t)NB * HID * 2);
  float*  gi   = (float*)alloc((size_t)NB * 3 * HID * 4);
  float*  gh   = (float*)alloc((size_t)NB * 3 * HID * 4);
  float*  accf = (float*)alloc((size_t)NB * HID * 4);
  float*  accb = (float*)alloc((size_t)NB * HID * 4);
  // total workspace ~131 MB  (fits comfortably in the 192 MB global L2)

  // 1) offsets + bf16 weight conversion + padded gathers
  offsets_kernel<<<1, 64, 0, stream>>>(a_sizes, p_sizes, a_off, p_off);
  const int nw = HID * HID, ng = 3 * HID * HID;
  f32_to_bf16_kernel<<<(nw + 255) / 256, 256, 0, stream>>>(Wq, Wqb, nw);
  f32_to_bf16_kernel<<<(nw + 255) / 256, 256, 0, stream>>>(Wk, Wkb, nw);
  f32_to_bf16_kernel<<<(nw + 255) / 256, 256, 0, stream>>>(Wv, Wvb, nw);
  f32_to_bf16_kernel<<<(nw + 255) / 256, 256, 0, stream>>>(Wo, Wob, nw);
  f32_to_bf16_kernel<<<(ng + 255) / 256, 256, 0, stream>>>(Wih_f, Wihf, ng);
  f32_to_bf16_kernel<<<(ng + 255) / 256, 256, 0, stream>>>(Whh_f, Whhf, ng);
  f32_to_bf16_kernel<<<(ng + 255) / 256, 256, 0, stream>>>(Wih_b, Wihb, ng);
  f32_to_bf16_kernel<<<(ng + 255) / 256, 256, 0, stream>>>(Whh_b, Whhb, ng);
  pad_kernel<<<(int)(NA / 256), 256, 0, stream>>>(a_feat, a_off, a_sizes, a_pad, MAXA);
  pad_kernel<<<(int)(NP / 256), 256, 0, stream>>>(p_feat, p_off, p_sizes, p_pad, MAXP);

  // 2) Q/K/V projections (WMMA GEMMs, 16x64 tile per wave, bf16 out)
  {
    const int Ma = NB * MAXA, Mp = NB * MAXP, Ngq = HID / 64;   // 6 groups
    int tilesQ = (Ma / 16) * Ngq;          // 12288 waves
    gemm_bias_kernel<<<tilesQ / 4, 128, 0, stream>>>(
        a_pad, (long long)HID, Wqb, HID, bq, Ngq, tilesQ, Qb, nullptr, HID);
    int tilesP = (Mp / 16) * Ngq;          // 3072 waves
    gemm_bias_kernel<<<tilesP / 4, 128, 0, stream>>>(
        p_pad, (long long)HID, Wkb, HID, bk, Ngq, tilesP, Kbuf, nullptr, HID);
    gemm_bias_kernel<<<tilesP / 4, 128, 0, stream>>>(
        p_pad, (long long)HID, Wvb, HID, bv, Ngq, tilesP, Vbuf, nullptr, HID);
  }

  // 3) fused masked attention: 256*6*8 = 12288 waves, 4 waves/block
  attention_kernel<<<12288 / 4, 128, 0, stream>>>(Qb, Kbuf, Vbuf, a_sizes, p_sizes, Obuf);

  // 4) output projection + bias + residual -> H (bf16): 2048*6 waves
  gemm_wo_residual_kernel<<<12288 / 4, 128, 0, stream>>>(
      Obuf, Wob, bo, a_feat, a_off, a_sizes, Hbuf);

  // 5) h0 = max over time; init states + pooled accumulators
  h0_init_kernel<<<(NB * HID) / 256, 256, 0, stream>>>(Hbuf, sf, sfb, sb, sbb, accf, accb);

  // 6) GRU. Pool uses only t < p_sizes[b] <= 32, so forward needs 32 steps;
  //    backward scans all 128 (its last 32 outputs are the pooled ones).
  const int Ngg = (3 * HID) / 64;                   // 18 N-groups
  const int tilesG = (NB / 16) * Ngg;               // 288 waves -> 72 blocks
  for (int t = 0; t < 32; ++t) {
    gemm_bias_kernel<<<tilesG / 4, 128, 0, stream>>>(
        Hbuf + (size_t)t * HID, (long long)(MAXA * HID), Wihf, HID, bih_f,
        Ngg, tilesG, nullptr, gi, 3 * HID);
    gemm_bias_kernel<<<tilesG / 4, 128, 0, stream>>>(
        sfb, (long long)HID, Whhf, HID, bhh_f, Ngg, tilesG, nullptr, gh, 3 * HID);
    gru_gate_kernel<<<(NB * HID) / 256, 256, 0, stream>>>(gi, gh, sf, sfb, accf, p_sizes, t);
  }
  for (int s = 0; s < MAXA; ++s) {
    const int t = MAXA - 1 - s;                     // reversed-time input
    gemm_bias_kernel<<<tilesG / 4, 128, 0, stream>>>(
        Hbuf + (size_t)t * HID, (long long)(MAXA * HID), Wihb, HID, bih_b,
        Ngg, tilesG, nullptr, gi, 3 * HID);
    gemm_bias_kernel<<<tilesG / 4, 128, 0, stream>>>(
        sbb, (long long)HID, Whhb, HID, bhh_b, Ngg, tilesG, nullptr, gh, 3 * HID);
    gru_gate_kernel<<<(NB * HID) / 256, 256, 0, stream>>>(gi, gh, sb, sbb, accb, p_sizes, t);
  }

  // 7) pooled bidirectional embedding
  final_kernel<<<(NB * HID) / 256, 256, 0, stream>>>(accf, accb, p_sizes, out);
}
